// DownsampleResidualVectorQuantize_18597208392128
// MI455X (gfx1250) — compile-verified, matrix operand order checked by reading, not run on hardware
//
#include <hip/hip_runtime.h>
#include <math.h>

typedef __attribute__((ext_vector_type(16))) _Float16 v16h;
typedef __attribute__((ext_vector_type(8)))  float    v8f;
typedef __attribute__((ext_vector_type(4)))  int      v4i_t;
typedef _Float16 half_t;

#define CHUNK_M 8192
#define TM 64
#define TN 128
#define TK 32

#if defined(__HIP_DEVICE_COMPILE__) && \
    __has_builtin(__builtin_amdgcn_global_load_async_to_lds_b128) && \
    __has_builtin(__builtin_amdgcn_s_wait_asynccnt)
#define USE_ASYNC_LDS 1
#else
#define USE_ASYNC_LDS 0
#endif

enum { MODE_F16_GELU = 0, MODE_CT_STORE = 1, MODE_CT_GAMMA_ADD = 2, MODE_CT_TRANSCONV = 3 };

__device__ __forceinline__ void stage16(const half_t* g, half_t* l)
{
#if USE_ASYNC_LDS
    __builtin_amdgcn_global_load_async_to_lds_b128(
        (v4i_t __attribute__((address_space(1)))*)(g),
        (v4i_t __attribute__((address_space(3)))*)(l), 0, 0);
#else
    *(float4*)l = *(const float4*)g;
#endif
}

__device__ __forceinline__ void stage_wait()
{
#if USE_ASYNC_LDS
    __builtin_amdgcn_s_wait_asynccnt(0);
#endif
}

// ---------------------------------------------------------------------------
// Tiled WMMA GEMM: C[M,N] = A[M,K] (f16, row-major) * Bw[N,K]^T (f16, row-major)
// 256 threads = 8 waves (wave32). Block tile 64x128, waves 4(M) x 2(N).
// Each wave: 1 A-frag reused across 4 B-frags -> 4 v_wmma_f32_16x16x32_f16
// per K-step. LDS double-buffered; staging via GLOBAL_LOAD_ASYNC_TO_LDS_B128
// (ASYNCcnt) when available.
// ---------------------------------------------------------------------------
template<int MODE>
__global__ __launch_bounds__(256)
void gemm_wmma_kernel(const half_t* __restrict__ A, const half_t* __restrict__ Bw,
                      const float* __restrict__ bias, const float* __restrict__ gamma,
                      float* __restrict__ outF, half_t* __restrict__ outH,
                      int N, int K, int m0, int Ttok, int Tout, int tmul, int toff)
{
    __shared__ half_t sA[2][TM][TK];
    __shared__ half_t sB[2][TN][TK];
    const int tid  = threadIdx.x;
    const int wave = tid >> 5;
    const int lane = tid & 31;
    const int r  = lane & 15;
    const int hi = lane >> 4;
    const int bm = blockIdx.x * TM;
    const int bn = blockIdx.y * TN;
    const int wm = (wave & 3) * 16;   // 4 waves along M (64)
    const int wn = (wave >> 2) * 64;  // 2 waves along N, 64 cols each
    v8f acc[4];
#pragma unroll
    for (int j = 0; j < 4; ++j) acc[j] = (v8f){0.f,0.f,0.f,0.f,0.f,0.f,0.f,0.f};

    // staging map: A tile 64x32 -> 1 x b128/thread; B tile 128x32 -> 2 x b128/thread
    const int arow = tid >> 2, acol = (tid & 3) * 8;
    const int brow = tid >> 1, bcol = (tid & 1) * 16;

    const int nt = K / TK;
    // preload tile 0 into buffer 0
    {
        const half_t* ga = A  + (size_t)(bm + arow) * K + acol;
        const half_t* gb = Bw + (size_t)(bn + brow) * K + bcol;
        stage16(ga, &sA[0][arow][acol]);
        stage16(gb,     &sB[0][brow][bcol]);
        stage16(gb + 8, &sB[0][brow][bcol + 8]);
    }
    stage_wait();
    __syncthreads();

    for (int it = 0; it < nt; ++it) {
        const int p = it & 1;
        if (it + 1 < nt) {
            const int k1 = (it + 1) * TK;
            const half_t* ga = A  + (size_t)(bm + arow) * K + k1 + acol;
            const half_t* gb = Bw + (size_t)(bn + brow) * K + k1 + bcol;
            stage16(ga, &sA[p ^ 1][arow][acol]);
            stage16(gb,     &sB[p ^ 1][brow][bcol]);
            stage16(gb + 8, &sB[p ^ 1][brow][bcol + 8]);
        }
        v16h fa, fb;
#pragma unroll
        for (int v = 0; v < 8; ++v) {
            // 16-bit A layout: lanes 0-15 hold K {0..7,16..23}, lanes 16-31 hold K {8..15,24..31}
            int kk = ((v < 4) ? (2 * v) : (16 + 2 * (v - 4))) + hi * 8;
            fa[2*v]   = sA[p][wm + r][kk];
            fa[2*v+1] = sA[p][wm + r][kk + 1];
        }
#pragma unroll
        for (int j = 0; j < 4; ++j) {
#pragma unroll
            for (int v = 0; v < 8; ++v) {
                int kk = ((v < 4) ? (2 * v) : (16 + 2 * (v - 4))) + hi * 8;
                fb[2*v]   = sB[p][wn + j * 16 + r][kk];
                fb[2*v+1] = sB[p][wn + j * 16 + r][kk + 1];
            }
            acc[j] = __builtin_amdgcn_wmma_f32_16x16x32_f16(false, fa, false, fb,
                                                            (short)0, acc[j], false, false);
        }
        stage_wait();      // this wave's async stores into buffer p^1 have landed
        __syncthreads();   // all waves done reading p / writing p^1
    }

    // Epilogue: D layout -> VGPR v, lane: M = v + 8*hi, N = lane&15
#pragma unroll
    for (int j = 0; j < 4; ++j) {
        int n = bn + wn + j * 16 + r;
        float bv = bias[n];
#pragma unroll
        for (int v = 0; v < 8; ++v) {
            int m = bm + wm + hi * 8 + v;
            float val = acc[j][v] + bv;
            if (MODE == MODE_F16_GELU) {
                float g = 0.5f * val * (1.0f + erff(val * 0.70710678118654752f));
                outH[(size_t)m * N + n] = (half_t)g;
            } else {
                int gm = m0 + m;
                int b = gm / Ttok, t = gm - b * Ttok;
                size_t oi = ((size_t)b * N + n) * (size_t)Tout + (size_t)t * tmul + toff;
                if (MODE == MODE_CT_STORE)          outF[oi]  = val;
                else if (MODE == MODE_CT_GAMMA_ADD) outF[oi] += gamma[n] * val;
                else                                 outF[oi]  = val;  // TRANSCONV
            }
        }
    }
}

// ---------------------------------------------------------------------------
// Fused depthwise causal conv (k=7, pad=6) + LayerNorm over channels,
// one block (128 threads) per token, emits f16 token-major A matrix.
// ---------------------------------------------------------------------------
__global__ __launch_bounds__(128)
void dwconv_ln_kernel(const float* __restrict__ X, const float* __restrict__ dww,
                      const float* __restrict__ dwb, const float* __restrict__ lng,
                      const float* __restrict__ lnb, half_t* __restrict__ A16,
                      int T, int m0)
{
    const int C = 512;
    int gm = m0 + blockIdx.x;
    int b = gm / T, t = gm - b * T;
    float vals[4];
    float s = 0.f, s2 = 0.f;
#pragma unroll
    for (int j = 0; j < 4; ++j) {
        int c = threadIdx.x + j * 128;
        const float* xr = X + ((size_t)b * C + c) * T;
        float acc = dwb[c];
#pragma unroll
        for (int kk = 0; kk < 7; ++kk) {
            int tt = t - 6 + kk;
            float xv = (tt >= 0) ? xr[tt] : 0.f;
            acc += dww[c * 7 + kk] * xv;
        }
        vals[j] = acc;
        s  += acc;
        s2 += acc * acc;
    }
    __shared__ float red[128], red2[128];
    red[threadIdx.x] = s; red2[threadIdx.x] = s2;
    __syncthreads();
    for (int off = 64; off > 0; off >>= 1) {
        if (threadIdx.x < off) {
            red[threadIdx.x]  += red[threadIdx.x + off];
            red2[threadIdx.x] += red2[threadIdx.x + off];
        }
        __syncthreads();
    }
    float mu  = red[0] * (1.f / 512.f);
    float var = red2[0] * (1.f / 512.f) - mu * mu;
    float rs  = rsqrtf(var + 1e-6f);
#pragma unroll
    for (int j = 0; j < 4; ++j) {
        int c = threadIdx.x + j * 128;
        float yv = (vals[j] - mu) * rs * lng[c] + lnb[c];
        A16[(size_t)blockIdx.x * C + c] = (half_t)yv;
    }
}

// A[m][2c+k] = X[b][c][2t+k]  (stride-2 k=2 causal conv has pad=0, extra=0)
__global__ void pack_conv_a_kernel(const float* __restrict__ X, half_t* __restrict__ A16,
                                   int Tin, int Tout, int m0, int total)
{
    int i = blockIdx.x * blockDim.x + threadIdx.x;
    if (i >= total) return;
    int c = i & 511;
    int m = i >> 9;
    int gm = m0 + m;
    int b = gm / Tout, t = gm - b * Tout;
    const float* src = X + ((size_t)b * 512 + c) * Tin + 2 * t;
    A16[(size_t)m * 1024 + 2 * c]     = (half_t)src[0];
    A16[(size_t)m * 1024 + 2 * c + 1] = (half_t)src[1];
}

// A[m][c] = X[b][c][t]  (channel-major -> token-major transpose + f16 cast)
__global__ void pack_t_kernel(const float* __restrict__ X, half_t* __restrict__ A16,
                              int T, int m0, int total)
{
    int i = blockIdx.x * blockDim.x + threadIdx.x;
    if (i >= total) return;
    int c = i & 511;
    int m = i >> 9;
    int gm = m0 + m;
    int b = gm / T, t = gm - b * T;
    A16[(size_t)m * 512 + c] = (half_t)X[((size_t)b * 512 + c) * T + t];
}

__global__ void cvt_f2h_kernel(const float* __restrict__ s, half_t* __restrict__ d, int n)
{
    int i = blockIdx.x * blockDim.x + threadIdx.x;
    if (i < n) d[i] = (half_t)s[i];
}

// wk[k][o][i] = w[i][o][k]   (trans_conv weight (I,O,K) -> per-tap (O,I) f16)
__global__ void pack_tconv_w_kernel(const float* __restrict__ w, half_t* __restrict__ wk, int n)
{
    int i = blockIdx.x * blockDim.x + threadIdx.x;
    if (i >= n) return;
    int k   = i >> 18;
    int rem = i & 262143;
    int o   = rem >> 9;
    int ii  = rem & 511;
    wk[i] = (half_t)w[((size_t)ii * 512 + o) * 2 + k];
}

// ---------------------------------------------------------------------------
// VQ stage kernels (8-dim codes: skinny, kept on VALU)
// ---------------------------------------------------------------------------
__global__ void vq_ze_kernel(const float* __restrict__ resid, const float* __restrict__ inw,
                             const float* __restrict__ inb, float* __restrict__ ze, int Tq)
{
    int i = blockIdx.x * blockDim.x + threadIdx.x;  // B*8*Tq
    if (i >= 8 * 8 * Tq) return;
    int t = i % Tq;
    int c = (i / Tq) & 7;
    int b = i / (8 * Tq);
    const float* rr = resid + (size_t)b * 512 * Tq + t;
    const float* wr = inw + c * 512;
    float acc = inb[c];
    for (int d = 0; d < 512; ++d) acc += wr[d] * rr[(size_t)d * Tq];
    ze[i] = acc;
}

__global__ __launch_bounds__(256)
void vq_argmax_kernel(const float* __restrict__ ze, const float* __restrict__ cb,
                      int* __restrict__ codes_out, int* __restrict__ idx_out,
                      float* __restrict__ lossAcc, int Tq, int stage)
{
    int lane  = threadIdx.x & 31;
    int token = blockIdx.x * 8 + (threadIdx.x >> 5);  // one wave per token
    int b = token / Tq, t = token - b * Tq;
    float e[8];
#pragma unroll
    for (int c = 0; c < 8; ++c) e[c] = ze[((size_t)b * 8 + c) * Tq + t];
    // argmax over cosine sim == argmax over dot(e, c)/|c|  (1/|e| is invariant)
    float bestS = -1e30f; int bestK = 0x7fffffff;
    for (int k = lane; k < 1024; k += 32) {
        const float* cr = cb + k * 8;
        float dot = 0.f, nn = 0.f;
#pragma unroll
        for (int c = 0; c < 8; ++c) { dot += e[c] * cr[c]; nn += cr[c] * cr[c]; }
        float sim = dot / fmaxf(sqrtf(nn), 1e-12f);
        if (sim > bestS || (sim == bestS && k < bestK)) { bestS = sim; bestK = k; }
    }
    for (int off = 16; off > 0; off >>= 1) {
        float os = __shfl_down(bestS, off, 32);
        int   ok = __shfl_down(bestK, off, 32);
        if (os > bestS || (os == bestS && ok < bestK)) { bestS = os; bestK = ok; }
    }
    bestK = __shfl(bestK, 0, 32);
    if (lane == 0) {
        idx_out[token] = bestK;
        codes_out[((size_t)b * 9 + stage) * Tq + t] = bestK;
        const float* cr = cb + bestK * 8;
        float se = 0.f;
#pragma unroll
        for (int c = 0; c < 8; ++c) { float d = e[c] - cr[c]; se += d * d; }
        atomicAdd(lossAcc, se);  // commit == cbl in forward value
    }
}

__global__ void vq_out_kernel(const float* __restrict__ cb, const int* __restrict__ idx,
                              const float* __restrict__ outw, const float* __restrict__ outb,
                              float* __restrict__ zq, float* __restrict__ resid, int Tq)
{
    int i = blockIdx.x * blockDim.x + threadIdx.x;  // B*512*Tq; i == (b*512+d)*Tq+t
    if (i >= 8 * 512 * Tq) return;
    int t = i % Tq;
    int d = (i / Tq) & 511;
    int b = i / (512 * Tq);
    int k = idx[b * Tq + t];
    const float* cr = cb + k * 8;
    const float* wr = outw + d * 8;
    float acc = outb[d];
#pragma unroll
    for (int c = 0; c < 8; ++c) acc += wr[c] * cr[c];
    zq[i]    += acc;
    resid[i] -= acc;
}

// ---------------------------------------------------------------------------
__global__ void copy_f32_kernel(const float* __restrict__ s, float* __restrict__ d, size_t n)
{
    size_t i = (size_t)blockIdx.x * blockDim.x + threadIdx.x;
    if (i < n) d[i] = s[i];
}
__global__ void zero_f32_kernel(float* __restrict__ d, size_t n)
{
    size_t i = (size_t)blockIdx.x * blockDim.x + threadIdx.x;
    if (i < n) d[i] = 0.f;
}
__global__ void finalize_loss_kernel(const float* __restrict__ acc,
                                     float* __restrict__ oc, float* __restrict__ ocb, float scale)
{
    if (blockIdx.x == 0 && threadIdx.x == 0) {
        float v = acc[0] * scale;
        *oc = v; *ocb = v;
    }
}

// ---------------------------------------------------------------------------
static inline int idiv_up(int a, int b) { return (a + b - 1) / b; }

extern "C" void kernel_launch(void* const* d_in, const int* in_sizes, int n_in,
                              void* d_out, int out_size, void* d_ws, size_t ws_size,
                              hipStream_t stream)
{
    (void)in_sizes; (void)n_in; (void)out_size; (void)ws_size;
    const int B = 8;
    const float* z = (const float*)d_in[0];
    auto P = [&](int i) { return (const float*)d_in[i]; };
    // Input index map (setup_inputs dict order):
    //   0 z; down0@1, down1@12, up0@23, up1@34 (each: conv_w, conv_b, dw_w, dw_b,
    //   ln_g, ln_b, w1, b1, w2, b2, gamma); vq s@45+5s (in_w,in_b,out_w,out_b,codebook); 90 n_q

    // ---- workspace carve ----
    char* wp = (char*)d_ws;
    auto alloc = [&](size_t bytes) { void* p = wp; wp += (bytes + 255) & ~(size_t)255; return p; };
    float*  buf0  = (float*)alloc((size_t)16777216 * 4);  // (8,512,4096)
    float*  buf1  = (float*)alloc((size_t)8388608  * 4);  // (8,512,2048)
    float*  buf2  = (float*)alloc((size_t)4194304  * 4);  // (8,512,1024)
    float*  zq    = (float*)alloc((size_t)4194304  * 4);
    float*  resid = (float*)alloc((size_t)4194304  * 4);
    half_t* A16   = (half_t*)alloc((size_t)CHUNK_M * 1024 * 2);
    half_t* G16   = (half_t*)alloc((size_t)CHUNK_M * 2048 * 2);
    half_t* W1h   = (half_t*)alloc((size_t)2048 * 512 * 2);
    half_t* W2h   = (half_t*)alloc((size_t)2048 * 512 * 2);
    half_t* Wch   = (half_t*)alloc((size_t)512 * 1024 * 2);
    float*  ze    = (float*)alloc((size_t)65536 * 4);
    int*    idxb  = (int*)alloc((size_t)8192 * 4);
    float*  lossA = (float*)alloc(256);

    auto cvt = [&](const float* s, half_t* d, int n) {
        cvt_f2h_kernel<<<idiv_up(n, 256), 256, 0, stream>>>(s, d, n);
    };
    auto gemm = [&](int mode, const half_t* A, const half_t* Bw, const float* bias,
                    const float* gamma, float* oF, half_t* oH,
                    int Mc, int N, int K, int m0, int Ttok, int Tout, int tmul, int toff) {
        dim3 g(Mc / TM, N / TN);
        if (mode == MODE_F16_GELU)
            gemm_wmma_kernel<MODE_F16_GELU><<<g, 256, 0, stream>>>(A, Bw, bias, gamma, oF, oH, N, K, m0, Ttok, Tout, tmul, toff);
        else if (mode == MODE_CT_STORE)
            gemm_wmma_kernel<MODE_CT_STORE><<<g, 256, 0, stream>>>(A, Bw, bias, gamma, oF, oH, N, K, m0, Ttok, Tout, tmul, toff);
        else if (mode == MODE_CT_GAMMA_ADD)
            gemm_wmma_kernel<MODE_CT_GAMMA_ADD><<<g, 256, 0, stream>>>(A, Bw, bias, gamma, oF, oH, N, K, m0, Ttok, Tout, tmul, toff);
        else
            gemm_wmma_kernel<MODE_CT_TRANSCONV><<<g, 256, 0, stream>>>(A, Bw, bias, gamma, oF, oH, N, K, m0, Ttok, Tout, tmul, toff);
    };
    auto convnext = [&](float* X, int T, int base) {
        cvt(P(base + 6), W1h, 2048 * 512);
        cvt(P(base + 8), W2h, 512 * 2048);
        int M = B * T;
        for (int m0 = 0; m0 < M; m0 += CHUNK_M) {
            dwconv_ln_kernel<<<CHUNK_M, 128, 0, stream>>>(X, P(base + 2), P(base + 3),
                                                          P(base + 4), P(base + 5), A16, T, m0);
            gemm(MODE_F16_GELU, A16, W1h, P(base + 7), nullptr, nullptr, G16,
                 CHUNK_M, 2048, 512, m0, T, T, 1, 0);
            gemm(MODE_CT_GAMMA_ADD, G16, W2h, P(base + 9), P(base + 10), X, nullptr,
                 CHUNK_M, 512, 2048, m0, T, T, 1, 0);
        }
    };
    auto downconv = [&](const float* Xin, int Tin, float* Xout, int base) {
        cvt(P(base + 0), Wch, 512 * 1024);  // (O,I,K) rows == [2c+k] pack order
        int Tout = Tin / 2, M = B * Tout;
        for (int m0 = 0; m0 < M; m0 += CHUNK_M) {
            int total = CHUNK_M * 512;
            pack_conv_a_kernel<<<idiv_up(total, 256), 256, 0, stream>>>(Xin, A16, Tin, Tout, m0, total);
            gemm(MODE_CT_STORE, A16, Wch, P(base + 1), nullptr, Xout, nullptr,
                 CHUNK_M, 512, 1024, m0, Tout, Tout, 1, 0);
        }
    };
    auto transconv = [&](const float* Xin, int Tin, float* Xout, int base) {
        pack_tconv_w_kernel<<<idiv_up(2 * 512 * 512, 256), 256, 0, stream>>>(P(base + 0), Wch, 2 * 512 * 512);
        int M = B * Tin;
        for (int m0 = 0; m0 < M; m0 += CHUNK_M) {
            int total = CHUNK_M * 512;
            pack_t_kernel<<<idiv_up(total, 256), 256, 0, stream>>>(Xin, A16, Tin, m0, total);
            for (int k = 0; k < 2; ++k)
                gemm(MODE_CT_TRANSCONV, A16, Wch + (size_t)k * 512 * 512, P(base + 1), nullptr,
                     Xout, nullptr, CHUNK_M, 512, 512, m0, Tin, Tin * 2, 2, k);
        }
    };

    // ---- encoder ----
    downconv(z, 4096, buf1, 1);
    convnext(buf1, 2048, 1);
    downconv(buf1, 2048, buf2, 12);
    convnext(buf2, 1024, 12);

    // ---- residual VQ (9 stages) ----
    const int Tq = 1024;
    const size_t nh = (size_t)B * 512 * Tq;
    copy_f32_kernel<<<idiv_up((int)nh, 256), 256, 0, stream>>>(buf2, resid, nh);
    zero_f32_kernel<<<idiv_up((int)nh, 256), 256, 0, stream>>>(zq, nh);
    zero_f32_kernel<<<1, 64, 0, stream>>>(lossA, 1);
    float* out_f     = (float*)d_out;
    int*   codes_out = (int*)(out_f + (size_t)B * 512 * 4096);
    for (int s = 0; s < 9; ++s) {
        int vb = 45 + s * 5;
        vq_ze_kernel<<<idiv_up(B * 8 * Tq, 256), 256, 0, stream>>>(resid, P(vb + 0), P(vb + 1), ze, Tq);
        vq_argmax_kernel<<<idiv_up(B * Tq * 32, 256), 256, 0, stream>>>(ze, P(vb + 4), codes_out,
                                                                        idxb, lossA, Tq, s);
        vq_out_kernel<<<idiv_up(B * 512 * Tq, 256), 256, 0, stream>>>(P(vb + 4), idxb, P(vb + 2),
                                                                      P(vb + 3), zq, resid, Tq);
    }

    // ---- decoder ----
    transconv(zq, 1024, buf1, 23);
    convnext(buf1, 2048, 23);
    transconv(buf1, 2048, buf0, 34);
    convnext(buf0, 4096, 34);

    // ---- outputs: out f32 | codes int32 | commit | cbl ----
    copy_f32_kernel<<<idiv_up(16777216, 256), 256, 0, stream>>>(buf0, out_f, (size_t)16777216);
    float* loss_out = (float*)(codes_out + (size_t)B * 9 * Tq);
    finalize_loss_kernel<<<1, 1, 0, stream>>>(lossA, loss_out, loss_out + 1,
                                              1.0f / (float)(B * 8 * Tq));
}
